// Summer_14826227106396
// MI455X (gfx1250) — compile-verified, matrix operand order checked by reading, not run on hardware
//
#include <hip/hip_runtime.h>
#include <hip/hip_bf16.h>
#include <stdint.h>

// Problem constants (match reference)
#define Bq    4
#define Nseq  4096
#define Dm    1024
#define Mrows (Bq * Nseq)   // 16384
#define D2m   (2 * Dm)      // 2048
#define D4m   (4 * Dm)      // 4096

typedef __bf16 bf16x16 __attribute__((ext_vector_type(16)));
typedef float  f32x8   __attribute__((ext_vector_type(8)));

union Frag { bf16x16 v; uint4 q[2]; };

__device__ __forceinline__ unsigned short f2bf(float f) {
  union { float f; uint32_t u; } c; c.f = f;
  uint32_t u = c.u;
  u += 0x7FFFu + ((u >> 16) & 1u);      // round-to-nearest-even
  return (unsigned short)(u >> 16);
}
__device__ __forceinline__ float bf2f(unsigned short h) {
  union { uint32_t u; float f; } c; c.u = ((uint32_t)h) << 16;
  return c.f;
}
__device__ __forceinline__ float silu_f(float x) { return x / (1.0f + __expf(-x)); }

// CDNA5 async global->LDS copy (ASYNCcnt-tracked, no VGPR round trip).
// lds_addr: workgroup-relative LDS byte address (VDST); gaddr: 64-bit global addr.
__device__ __forceinline__ void async_g2l_b128(uint32_t lds_addr, const void* gaddr) {
  asm volatile("global_load_async_to_lds_b128 %0, %1, off"
               :: "v"(lds_addr), "v"((unsigned long long)(uintptr_t)gaddr)
               : "memory");
}
__device__ __forceinline__ void wait_async0() {
  asm volatile("s_wait_asynccnt 0x0" ::: "memory");
}
__device__ __forceinline__ uint32_t lds_byte_addr(const void* p) {
  // AS3 -> generic cast places the LDS byte offset in the low 32 bits.
  return (uint32_t)(uintptr_t)p;
}

// ---------------------------------------------------------------------------
// Generic bf16 WMMA GEMM:  C[M x Nout] = scale * (A[M x K] @ Bt[Nout x K]^T)
//                                        (+ Cin) (+ bias) (silu?) -> f32/bf16
// A rows may be shifted within each batch segment (for the 3-tap conv),
// out-of-range rows read as zero.
// Block: 256 threads (8 waves), tile 128x128, K-step 32, double-buffered LDS,
// tiles staged with global_load_async_to_lds_b128 (one 16B chunk per lane).
// ---------------------------------------------------------------------------
#define BM  128
#define BN  128
#define BK  32
#define BKP 40   // padded row stride (elements) -> 80B rows, conflict-friendly

__global__ __launch_bounds__(256) void gemm_bf16_kernel(
    const unsigned short* __restrict__ A, int lda,
    const unsigned short* __restrict__ Bt, int K, int Nout,
    const float* __restrict__ bias, const float* __restrict__ Cin,
    float* __restrict__ Cf, unsigned short* __restrict__ Cb,
    int batchN, int shift, float scale, int act)
{
  __shared__ unsigned short As[2][BM][BKP];
  __shared__ unsigned short Bs[2][BN][BKP];

  const int tid   = threadIdx.x;
  const int lane  = tid & 31;
  const int wave  = tid >> 5;
  const int waveM = wave & 3;    // 0..3 -> 32-row strip
  const int waveN = wave >> 2;   // 0..1 -> 64-col strip
  const int rowBase = blockIdx.y * BM;
  const int colBase = blockIdx.x * BN;

  f32x8 acc[2][4];
#pragma unroll
  for (int mt = 0; mt < 2; ++mt)
#pragma unroll
    for (int nt = 0; nt < 4; ++nt)
#pragma unroll
      for (int e = 0; e < 8; ++e) acc[mt][nt][e] = 0.0f;

  const int nK = K >> 5;

  // Per-thread tile-load descriptors (k-independent; shift divide hoisted here).
  const unsigned short* aPtr[2];
  const unsigned short* bPtr[2];
  bool aOk[2];
  int  rowL[2], kcL[2];
#pragma unroll
  for (int i = 0; i < 2; ++i) {
    const int c   = tid + i * 256;     // 0..511 chunks of 8 bf16
    const int row = c >> 2;
    const int kc  = (c & 3) << 3;
    rowL[i] = row; kcL[i] = kc;
    int r = rowBase + row;
    int srcRow = r;
    bool ok = true;
    if (shift != 0) {
      int bb = r / batchN;
      int nn = r - bb * batchN + shift;
      ok     = (nn >= 0) && (nn < batchN);
      srcRow = bb * batchN + nn;
    }
    aOk[i]  = ok;
    aPtr[i] = A  + (size_t)srcRow * lda + kc;
    bPtr[i] = Bt + (size_t)(colBase + row) * K + kc;
  }

  auto issueTile = [&](int s, int kb) {
#pragma unroll
    for (int i = 0; i < 2; ++i) {
      if (aOk[i]) {
        async_g2l_b128(lds_byte_addr(&As[s][rowL[i]][kcL[i]]), aPtr[i] + kb);
      } else {
        uint4 z; z.x = 0u; z.y = 0u; z.z = 0u; z.w = 0u;
        *(uint4*)&As[s][rowL[i]][kcL[i]] = z;
      }
      async_g2l_b128(lds_byte_addr(&Bs[s][rowL[i]][kcL[i]]), bPtr[i] + kb);
    }
  };

  issueTile(0, 0);
  wait_async0();
  __syncthreads();

  const int lrow = lane & 15;            // A row / B col within 16x16 tile
  const int kh   = (lane >> 4) << 3;     // k-half: 0 or 8

  for (int ki = 0; ki < nK; ++ki) {
    const int s = ki & 1;
    if (ki + 1 < nK) issueTile(s ^ 1, (ki + 1) << 5);

    Frag afr[2], bfr[4];
#pragma unroll
    for (int mt = 0; mt < 2; ++mt) {
      int m = waveM * 32 + mt * 16 + lrow;
      afr[mt].q[0] = *(const uint4*)&As[s][m][kh];        // k = kh..kh+7
      afr[mt].q[1] = *(const uint4*)&As[s][m][kh + 16];   // k = 16+kh..
    }
#pragma unroll
    for (int nt = 0; nt < 4; ++nt) {
      int n = waveN * 64 + nt * 16 + lrow;
      bfr[nt].q[0] = *(const uint4*)&Bs[s][n][kh];
      bfr[nt].q[1] = *(const uint4*)&Bs[s][n][kh + 16];
    }
#pragma unroll
    for (int mt = 0; mt < 2; ++mt)
#pragma unroll
      for (int nt = 0; nt < 4; ++nt)
        acc[mt][nt] = __builtin_amdgcn_wmma_f32_16x16x32_bf16(
            false, afr[mt].v, false, bfr[nt].v, (short)0, acc[mt][nt], false, false);

    // Wait for next stage's async LDS writes only after compute (hides HBM latency),
    // then publish the tile to the workgroup.
    wait_async0();
    __syncthreads();
  }

  // Epilogue: C/D layout -> row = r + 8*(lane>=16), col = lane%16 per 16x16 tile
  const int rq = (lane >> 4) << 3;
#pragma unroll
  for (int mt = 0; mt < 2; ++mt) {
#pragma unroll
    for (int nt = 0; nt < 4; ++nt) {
      const int col = colBase + waveN * 64 + nt * 16 + lrow;
#pragma unroll
      for (int r = 0; r < 8; ++r) {
        const int row  = rowBase + waveM * 32 + mt * 16 + rq + r;
        const size_t idx = (size_t)row * Nout + col;
        float vv = acc[mt][nt][r] * scale;
        if (Cin)  vv += Cin[idx];
        if (bias) vv += bias[col];
        if (act)  vv = silu_f(vv);
        if (Cf) Cf[idx] = vv;
        if (Cb) Cb[idx] = f2bf(vv);
      }
    }
  }
}

// ---------------------------------------------------------------------------
// Elementwise / reduction kernels
// ---------------------------------------------------------------------------
__global__ void params_kernel(const float* __restrict__ nu_log,
                              const float* __restrict__ theta_log,
                              const float* __restrict__ gamma_log,
                              float* __restrict__ fr, float* __restrict__ fi,
                              float* __restrict__ gamma) {
  int d = blockIdx.x * blockDim.x + threadIdx.x;
  if (d < Dm) {
    float nu = expf(-expf(nu_log[d]));
    float th = expf(theta_log[d]);
    gamma[d] = expf(gamma_log[d]);
    fr[d] = nu * cosf(th);
    fi[d] = nu * sinf(th);
  }
}

__global__ void cast_bf16_kernel(const float* __restrict__ src,
                                 unsigned short* __restrict__ dst, int n) {
  int stride = gridDim.x * blockDim.x;
  for (int i = blockIdx.x * blockDim.x + threadIdx.x; i < n; i += stride)
    dst[i] = f2bf(src[i]);
}

// W [Krows x Ncols] (row-major) -> Wt [Ncols x Krows] bf16
__global__ void transpose_cast_kernel(const float* __restrict__ W,
                                      unsigned short* __restrict__ Wt,
                                      int Krows, int Ncols) {
  __shared__ float tile[32][33];
  const int bx = blockIdx.x * 32, by = blockIdx.y * 32;
  for (int i = threadIdx.y; i < 32; i += 8)
    tile[i][threadIdx.x] = W[(size_t)(by + i) * Ncols + bx + threadIdx.x];
  __syncthreads();
  for (int i = threadIdx.y; i < 32; i += 8)
    Wt[(size_t)(bx + i) * Krows + by + threadIdx.x] = f2bf(tile[threadIdx.x][i]);
}

// conv_w [co][ci][3] -> Wc[k][co][ci] bf16  (each tap: Bt layout [Nout=co][K=ci])
__global__ void conv_repack_kernel(const float* __restrict__ cw,
                                   unsigned short* __restrict__ Wc) {
  int stride = gridDim.x * blockDim.x;
  const int total = 3 * Dm * Dm;
  for (int i = blockIdx.x * blockDim.x + threadIdx.x; i < total; i += stride) {
    int k   = i / (Dm * Dm);
    int rem = i - k * Dm * Dm;
    int co  = rem / Dm;
    int ci  = rem - co * Dm;
    Wc[i] = f2bf(cw[((size_t)co * Dm + ci) * 3 + k]);
  }
}

// softmax over sequence axis + weighted sum with q: glob[b][d]
// s: [B][N][D] f32 logits;  u (bf16): q at stride D4m
__global__ __launch_bounds__(256) void softmax_glob_kernel(
    const float* __restrict__ s, const unsigned short* __restrict__ u,
    float* __restrict__ glob) {
  const int b = blockIdx.y;
  const int d = blockIdx.x * blockDim.x + threadIdx.x;
  const float* sp = s + (size_t)b * Nseq * Dm + d;
  float m = -3.4e38f;
  for (int n = 0; n < Nseq; ++n) m = fmaxf(m, sp[(size_t)n * Dm]);
  float z = 0.f;
  for (int n = 0; n < Nseq; ++n) z += expf(sp[(size_t)n * Dm] - m);
  const unsigned short* qp = u + (size_t)b * Nseq * D4m + d;
  float acc = 0.f;
  for (int n = 0; n < Nseq; ++n)
    acc += expf(sp[(size_t)n * Dm] - m) * bf2f(qp[(size_t)n * D4m]);
  glob[b * Dm + d] = acc / z;
}

// in-place: o-slot of u (bf16 at +Dm within each 4D row) *= glob[b][d]
__global__ void om_kernel(unsigned short* __restrict__ u,
                          const float* __restrict__ glob) {
  int stride = gridDim.x * blockDim.x;
  const int total = Mrows * Dm;
  for (int i = blockIdx.x * blockDim.x + threadIdx.x; i < total; i += stride) {
    int row = i / Dm;
    int d   = i - row * Dm;
    int b   = row / Nseq;
    size_t idx = (size_t)row * D4m + Dm + d;
    u[idx] = f2bf(bf2f(u[idx]) * glob[b * Dm + d]);
  }
}

// sequential complex LRU scan over N, in place on v [B][N][2D] f32
__global__ void scan_kernel(float* __restrict__ v, const float* __restrict__ fr,
                            const float* __restrict__ fi,
                            const float* __restrict__ gamma) {
  const int g = blockIdx.x * blockDim.x + threadIdx.x;   // 0..B*D-1
  const int b = g / Dm;
  const int d = g - b * Dm;
  const float fR = fr[d], fI = fi[d], ga = gamma[d];
  float hr = 0.f, hi = 0.f;
  size_t idx = (size_t)b * Nseq * D2m + d;
  for (int n = 0; n < Nseq; ++n) {
    const float ur = ga * v[idx];
    const float ui = ga * v[idx + Dm];
    const float nr = fR * hr - fI * hi + ur;
    const float ni = fR * hi + fI * hr + ui;
    hr = nr; hi = ni;
    v[idx] = hr; v[idx + Dm] = hi;
    idx += D2m;
  }
}

// h = [hr|hi] * silu(g); RMSNorm over 2D; -> bf16
__global__ __launch_bounds__(256) void gate_norm_kernel(
    const float* __restrict__ v, const unsigned short* __restrict__ u,
    unsigned short* __restrict__ hb) {
  const int row = blockIdx.x;
  const int tid = threadIdx.x;
  const float* hv = v + (size_t)row * D2m;
  const unsigned short* gp = u + (size_t)row * D4m + D2m;
  float t[8];
  float ss = 0.f;
#pragma unroll
  for (int i = 0; i < 8; ++i) {
    const int j = tid + i * 256;
    const float g = bf2f(gp[j]);
    const float val = hv[j] * (g / (1.f + __expf(-g)));
    t[i] = val; ss += val * val;
  }
  __shared__ float red[256];
  red[tid] = ss; __syncthreads();
  for (int st2 = 128; st2 > 0; st2 >>= 1) {
    if (tid < st2) red[tid] += red[tid + st2];
    __syncthreads();
  }
  const float rms = rsqrtf(red[0] * (1.0f / (float)D2m) + 1e-5f);
#pragma unroll
  for (int i = 0; i < 8; ++i)
    hb[(size_t)row * D2m + tid + i * 256] = f2bf(t[i] * rms);
}

// ---------------------------------------------------------------------------
extern "C" void kernel_launch(void* const* d_in, const int* in_sizes, int n_in,
                              void* d_out, int out_size, void* d_ws, size_t ws_size,
                              hipStream_t stream) {
  (void)in_sizes; (void)n_in; (void)out_size; (void)ws_size;

  const float* x         = (const float*)d_in[0];
  const float* W_in      = (const float*)d_in[1];
  const float* b_in      = (const float*)d_in[2];
  const float* W_mid     = (const float*)d_in[3];
  const float* b_mid     = (const float*)d_in[4];
  const float* W_out     = (const float*)d_in[5];
  const float* b_out     = (const float*)d_in[6];
  const float* W_from    = (const float*)d_in[7];
  const float* conv_w    = (const float*)d_in[8];
  const float* conv_b    = (const float*)d_in[9];
  const float* nu_log    = (const float*)d_in[10];
  const float* theta_log = (const float*)d_in[11];
  const float* gamma_log = (const float*)d_in[12];
  float* out = (float*)d_out;

  char* w = (char*)d_ws;
  size_t off = 0;
  auto alloc = [&](size_t bytes) -> void* {
    void* p = w + off;
    off = (off + bytes + 255) & ~(size_t)255;
    return p;
  };

  float*          fr    = (float*)alloc((size_t)Dm * 4);
  float*          fi    = (float*)alloc((size_t)Dm * 4);
  float*          gamma = (float*)alloc((size_t)Dm * 4);
  unsigned short* WtIn  = (unsigned short*)alloc((size_t)D4m * Dm * 2);
  unsigned short* WtMid = (unsigned short*)alloc((size_t)D2m * Dm * 2);
  unsigned short* WtOut = (unsigned short*)alloc((size_t)Dm * D2m * 2);
  unsigned short* WtFrm = (unsigned short*)alloc((size_t)Dm * Dm * 2);
  unsigned short* Wc    = (unsigned short*)alloc((size_t)3 * Dm * Dm * 2);
  unsigned short* xb    = (unsigned short*)alloc((size_t)Mrows * Dm * 2);
  unsigned short* ub    = (unsigned short*)alloc((size_t)Mrows * D4m * 2); // [q|o|g] bf16
  float*          cacc  = (float*)alloc((size_t)Mrows * Dm * 4);           // conv acc, reused as s
  unsigned short* cxf   = (unsigned short*)alloc((size_t)Mrows * Dm * 2);
  float*          glob  = (float*)alloc((size_t)Bq * Dm * 4);
  float*          vbuf  = (float*)alloc((size_t)Mrows * D2m * 4);
  unsigned short* hb    = (unsigned short*)alloc((size_t)Mrows * D2m * 2);

  // 1) LRU parameters
  params_kernel<<<(Dm + 255) / 256, 256, 0, stream>>>(nu_log, theta_log, gamma_log,
                                                      fr, fi, gamma);
  // 2) cast x -> bf16
  cast_bf16_kernel<<<4096, 256, 0, stream>>>(x, xb, Mrows * Dm);
  // 3) weight transposes/casts to Bt[Nout][K] bf16
  transpose_cast_kernel<<<dim3(D4m / 32, Dm / 32), dim3(32, 8), 0, stream>>>(W_in,  WtIn,  Dm,  D4m);
  transpose_cast_kernel<<<dim3(D2m / 32, Dm / 32), dim3(32, 8), 0, stream>>>(W_mid, WtMid, Dm,  D2m);
  transpose_cast_kernel<<<dim3(Dm / 32, D2m / 32), dim3(32, 8), 0, stream>>>(W_out, WtOut, D2m, Dm);
  transpose_cast_kernel<<<dim3(Dm / 32, Dm / 32),  dim3(32, 8), 0, stream>>>(W_from, WtFrm, Dm, Dm);
  conv_repack_kernel<<<2048, 256, 0, stream>>>(conv_w, Wc);

  auto gemm = [&](const unsigned short* A, int lda, const unsigned short* Bt, int K,
                  int Nout, const float* bias, const float* Cin, float* Cf,
                  unsigned short* Cb, int shift, float scale, int act) {
    dim3 grid(Nout / BN, Mrows / BM);
    gemm_bf16_kernel<<<grid, 256, 0, stream>>>(A, lda, Bt, K, Nout, bias, Cin, Cf, Cb,
                                               Nseq, shift, scale, act);
  };

  // 4) u = x @ W_in + b_in  -> bf16 [q|o|g]
  gemm(xb, Dm, WtIn, Dm, D4m, b_in, nullptr, nullptr, ub, 0, 1.0f, 0);

  // 5) conv: three shifted GEMMs on q (= ub, lda=4D), last tap: +bias, silu, bf16
  gemm(ub, D4m, Wc + 0 * Dm * Dm, Dm, Dm, nullptr, nullptr, cacc, nullptr, -1, 1.0f, 0);
  gemm(ub, D4m, Wc + 1 * (size_t)Dm * Dm, Dm, Dm, nullptr, cacc, cacc, nullptr, 0, 1.0f, 0);
  gemm(ub, D4m, Wc + 2 * (size_t)Dm * Dm, Dm, Dm, conv_b, cacc, nullptr, cxf, 1, 1.0f, 1);

  // 6) attention logits s = (cxf @ W_from) * 1/sqrt(D)   (reuse cacc as s)
  gemm(cxf, Dm, WtFrm, Dm, Dm, nullptr, nullptr, cacc, nullptr, 0, 0.03125f, 0);

  // 7) softmax over sequence + weighted q-sum -> glob[b][d]
  softmax_glob_kernel<<<dim3(Dm / 256, Bq), 256, 0, stream>>>(cacc, ub, glob);

  // 8) om = o * glob   (in place in ub's o-slot)
  om_kernel<<<4096, 256, 0, stream>>>(ub, glob);

  // 9) v = om @ W_mid + b_mid  -> f32
  gemm(ub + Dm, D4m, WtMid, Dm, D2m, b_mid, nullptr, vbuf, nullptr, 0, 1.0f, 0);

  // 10) complex LRU scan (in place, gamma applied to inputs)
  scan_kernel<<<(Bq * Dm) / 256, 256, 0, stream>>>(vbuf, fr, fi, gamma);

  // 11) gate by silu(g), RMSNorm -> bf16
  gate_norm_kernel<<<Mrows, 256, 0, stream>>>(vbuf, ub, hb);

  // 12) out = h @ W_out + b_out  -> f32 to d_out
  gemm(hb, D2m, WtOut, D2m, Dm, b_out, nullptr, out, nullptr, 0, 1.0f, 0);
}
